// MultiHeadAttention_14809047237155
// MI455X (gfx1250) — matrix operand order repaired and checked by reading, not compiled, not run on hardware
//
#include <hip/hip_runtime.h>

// MultiHeadAttention forward for MI455X (gfx1250, wave32, WMMA).
// - One-time f32->f16 conversion of activations/weights (halves GEMM traffic,
//   removes per-tile cvt from all inner loops).
// - Projection / output GEMMs: 64x128 block tile, 2x2 WMMA tiles per wave.
// - Fused flash-style attention: scores -> in-register softmax -> P@V; the
//   256MB attention matrix is written exactly once and never re-read.

typedef __attribute__((ext_vector_type(16))) _Float16 v16h;
typedef __attribute__((ext_vector_type(8)))  _Float16 v8h;
typedef __attribute__((ext_vector_type(8)))  float    v8f;

#define BK 32
#define LDP (BK + 8)   // +16B pad keeps 16B alignment for b128 LDS reads

union V16H { v16h v; v8h h[2]; };

// ---------------- f32 -> f16 conversion (one-time pre-pass) ----------------
__global__ void __launch_bounds__(256)
mha_cvt_f16_kernel(const float* __restrict__ src, _Float16* __restrict__ dst, int n)
{
    const int i = (blockIdx.x * 256 + threadIdx.x) * 4;
    if (i >= n) return;
    const float4 f = *reinterpret_cast<const float4*>(src + i);
    union { _Float16 h[4]; unsigned long long u; } t;
    t.h[0] = (_Float16)f.x; t.h[1] = (_Float16)f.y;
    t.h[2] = (_Float16)f.z; t.h[3] = (_Float16)f.w;
    *reinterpret_cast<unsigned long long*>(dst + i) = t.u;
}

// ---------------- f16 GEMM tile: C[64x128] per block, 2x2 tiles per wave ---
// A row-major [M][K] f16, B row-major [N][K] f16 (Linear weight layout).
// 256 threads = 8 waves in a 2x4 grid; each wave owns a 32x32 patch.
__device__ __forceinline__ void gemm_tile_f16x4(const _Float16* __restrict__ A, int lda,
                                                const _Float16* __restrict__ B, int ldb,
                                                int K, int m0, int n0, v8f acc[2][2])
{
    __shared__ _Float16 As[64][LDP];
    __shared__ _Float16 Bs[128][LDP];
    const int tid  = threadIdx.x;
    const int lane = tid & 31;
    const int wid  = tid >> 5;
    const int wm   = wid >> 2;            // 0..1
    const int wn   = wid & 3;             // 0..3
    const int nl   = lane & 15;
    const int ka   = (lane >> 4) << 3;    // A-frag k offset (0/8)
    const int kb   = (lane >> 4) << 3;    // B-frag k offset (0/8)

    for (int k0 = 0; k0 < K; k0 += BK) {
        {   // stage A (64x32 halfs): 8 halfs (16B) per thread
            const int r = tid >> 2;
            const int c = (tid & 3) << 3;
            *reinterpret_cast<v8h*>(&As[r][c]) =
                *reinterpret_cast<const v8h*>(A + (size_t)(m0 + r) * lda + k0 + c);
        }
        {   // stage B (128x32 halfs): 16 halfs (32B) per thread
            const int r = tid >> 1;
            const int c = (tid & 1) << 4;
            const _Float16* src = B + (size_t)(n0 + r) * ldb + k0 + c;
            *reinterpret_cast<v8h*>(&Bs[r][c])     = *reinterpret_cast<const v8h*>(src);
            *reinterpret_cast<v8h*>(&Bs[r][c + 8]) = *reinterpret_cast<const v8h*>(src + 8);
        }
        __syncthreads();
        V16H a[2], bb[2];
#pragma unroll
        for (int mi = 0; mi < 2; ++mi) {
            const int mA = wm * 32 + mi * 16 + nl;
            a[mi].h[0] = *reinterpret_cast<const v8h*>(&As[mA][ka]);
            a[mi].h[1] = *reinterpret_cast<const v8h*>(&As[mA][ka + 16]);
        }
#pragma unroll
        for (int ni = 0; ni < 2; ++ni) {
            const int nB = wn * 32 + ni * 16 + nl;
            bb[ni].h[0] = *reinterpret_cast<const v8h*>(&Bs[nB][kb]);
            bb[ni].h[1] = *reinterpret_cast<const v8h*>(&Bs[nB][kb + 16]);
        }
#pragma unroll
        for (int mi = 0; mi < 2; ++mi)
#pragma unroll
            for (int ni = 0; ni < 2; ++ni)
                acc[mi][ni] = __builtin_amdgcn_wmma_f32_16x16x32_f16(
                    false, a[mi].v, false, bb[ni].v, (short)0, acc[mi][ni], false, false);
        __syncthreads();
    }
}

// ---------------- Projection: heads = X @ W^T + b (f16 out) ----------------
// transpose_store==0: out[b][h][s][d] (q, k); ==1: out[b][h][d][s] (v).
__global__ void __launch_bounds__(256)
mha_proj_kernel(const _Float16* __restrict__ X, const _Float16* __restrict__ W,
                const float* __restrict__ bias, _Float16* __restrict__ out,
                int transpose_store)
{
    const int m0 = blockIdx.y * 64;
    const int n0 = blockIdx.x * 128;
    v8f acc[2][2] = {};
    gemm_tile_f16x4(X, 1024, W, 1024, 1024, m0, n0, acc);

    const int lane  = threadIdx.x & 31;
    const int wid   = threadIdx.x >> 5;
    const int mbase = (lane >> 4) << 3;
    const int nl    = lane & 15;
#pragma unroll
    for (int mi = 0; mi < 2; ++mi) {
#pragma unroll
        for (int ni = 0; ni < 2; ++ni) {
            const int n = n0 + (wid & 3) * 32 + ni * 16 + nl;
            const int h = n >> 6, d = n & 63;
            const float bv = bias[n];
#pragma unroll
            for (int r = 0; r < 8; ++r) {
                const int m = m0 + (wid >> 2) * 32 + mi * 16 + mbase + r;
                const int b = m >> 10, s = m & 1023;
                const _Float16 val = (_Float16)(acc[mi][ni][r] + bv);
                if (!transpose_store)
                    out[(((size_t)(b * 16 + h)) * 1024 + s) * 64 + d] = val;
                else
                    out[(((size_t)(b * 16 + h)) * 64 + d) * 1024 + s] = val;
            }
        }
    }
}

// ---------------- Fused attention: scores -> softmax -> P@V ----------------
// One block = 16 query rows of one (b,h). 8 waves; wave w owns key columns
// [128w, 128w+128) for scores/softmax and the same K-slice of P@V.
__global__ void __launch_bounds__(256)
mha_fused_attn_kernel(const _Float16* __restrict__ q,   // [b,h,s,d] f16
                      const _Float16* __restrict__ k,   // [b,h,s,d] f16
                      const _Float16* __restrict__ v,   // [b,h,d,s] f16
                      const unsigned char* __restrict__ mask, // [b,s,s]
                      float* __restrict__ attn,         // [b,h,s,s] probs (f32)
                      _Float16* __restrict__ ctx)       // [b*s, h*64] f16
{
    const int z  = blockIdx.y;            // b*16 + h
    const int b  = z >> 4;
    const int h  = z & 15;
    const int m0 = blockIdx.x * 16;       // query row block

    const _Float16* qp = q + (size_t)z * 1024 * 64;
    const _Float16* kp = k + (size_t)z * 1024 * 64;
    const _Float16* vp = v + (size_t)z * 64 * 1024;
    float* attnz       = attn + (size_t)z * 1024 * 1024;

    __shared__ _Float16 Qs[16][72];            // q tile, padded
    __shared__ _Float16 Pbuf[16][1024 + 8];    // probabilities, f16
    __shared__ float    rowred[16][8];         // per-wave row partials
    __shared__ float    ctxred[8][16][64];     // per-wave P@V partials

    const int tid   = threadIdx.x;
    const int lane  = tid & 31;
    const int w     = tid >> 5;
    const int nlane = lane & 15;
    const int mbase = (lane >> 4) << 3;
    const int ka    = (lane >> 4) << 3;
    const int kb    = (lane >> 4) << 3;

    // Stage q tile (16x64 halfs): 4 halfs (8B) per thread.
    {
        const int r = tid >> 4;
        const int c = (tid & 15) << 2;
        *reinterpret_cast<unsigned long long*>(&Qs[r][c]) =
            *reinterpret_cast<const unsigned long long*>(qp + (size_t)(m0 + r) * 64 + c);
    }
    __syncthreads();

    // A fragments for both k-steps (K = 64), shared across 8 column tiles.
    V16H a0, a1;
    a0.h[0] = *reinterpret_cast<const v8h*>(&Qs[nlane][ka]);
    a0.h[1] = *reinterpret_cast<const v8h*>(&Qs[nlane][ka + 16]);
    a1.h[0] = *reinterpret_cast<const v8h*>(&Qs[nlane][32 + ka]);
    a1.h[1] = *reinterpret_cast<const v8h*>(&Qs[nlane][48 + ka]);

    // ---- Phase 1: scores for this wave's 128 key columns ----
    v8f sacc[8];
#pragma unroll
    for (int t = 0; t < 8; ++t) sacc[t] = (v8f){};

#pragma unroll
    for (int t = 0; t < 8; ++t) {
        const int ncol = w * 128 + t * 16 + nlane;     // key row index
        const _Float16* krow = kp + (size_t)ncol * 64 + kb;
        V16H b0, b1;
        b0.h[0] = *reinterpret_cast<const v8h*>(krow);
        b0.h[1] = *reinterpret_cast<const v8h*>(krow + 16);
        b1.h[0] = *reinterpret_cast<const v8h*>(krow + 32);
        b1.h[1] = *reinterpret_cast<const v8h*>(krow + 48);
        sacc[t] = __builtin_amdgcn_wmma_f32_16x16x32_f16(
            false, a0.v, false, b0.v, (short)0, sacc[t], false, false);
        sacc[t] = __builtin_amdgcn_wmma_f32_16x16x32_f16(
            false, a1.v, false, b1.v, (short)0, sacc[t], false, false);
    }

    // scale + mask
#pragma unroll
    for (int t = 0; t < 8; ++t) {
        const int ncol = w * 128 + t * 16 + nlane;
#pragma unroll
        for (int r = 0; r < 8; ++r) {
            float val = sacc[t][r] * 0.125f;
            const int m = m0 + mbase + r;
            if (mask[((size_t)b * 1024 + m) * 1024 + ncol]) val = -1e9f;
            sacc[t][r] = val;
        }
    }

    // ---- Softmax over each of the 16 rows (1024 columns) ----
    float lm[8];
#pragma unroll
    for (int r = 0; r < 8; ++r) {
        float m = sacc[0][r];
#pragma unroll
        for (int t = 1; t < 8; ++t) m = fmaxf(m, sacc[t][r]);
        for (int off = 1; off < 16; off <<= 1)   // xor<=8 stays inside row half
            m = fmaxf(m, __shfl_xor(m, off));
        lm[r] = m;
    }
    if (nlane == 0) {
#pragma unroll
        for (int r = 0; r < 8; ++r) rowred[mbase + r][w] = lm[r];
    }
    __syncthreads();
    float rowmax[8];
#pragma unroll
    for (int r = 0; r < 8; ++r) {
        float m = rowred[mbase + r][0];
#pragma unroll
        for (int ww = 1; ww < 8; ++ww) m = fmaxf(m, rowred[mbase + r][ww]);
        rowmax[r] = m;
    }
    __syncthreads();   // rowred reused for sums

    float ls[8];
#pragma unroll
    for (int r = 0; r < 8; ++r) ls[r] = 0.0f;
#pragma unroll
    for (int t = 0; t < 8; ++t) {
#pragma unroll
        for (int r = 0; r < 8; ++r) {
            const float e = __expf(sacc[t][r] - rowmax[r]);
            sacc[t][r] = e;
            ls[r] += e;
        }
    }
#pragma unroll
    for (int r = 0; r < 8; ++r) {
        float s = ls[r];
        for (int off = 1; off < 16; off <<= 1) s += __shfl_xor(s, off);
        ls[r] = s;
    }
    if (nlane == 0) {
#pragma unroll
        for (int r = 0; r < 8; ++r) rowred[mbase + r][w] = ls[r];
    }
    __syncthreads();
    float rinv[8];
#pragma unroll
    for (int r = 0; r < 8; ++r) {
        float s = rowred[mbase + r][0];
#pragma unroll
        for (int ww = 1; ww < 8; ++ww) s += rowred[mbase + r][ww];
        rinv[r] = 1.0f / s;
    }

    // normalize; write probs once to global (required output) and to LDS f16
#pragma unroll
    for (int t = 0; t < 8; ++t) {
        const int ncol = w * 128 + t * 16 + nlane;
#pragma unroll
        for (int r = 0; r < 8; ++r) {
            const float pval = sacc[t][r] * rinv[r];
            const int ml = mbase + r;
            attnz[(size_t)(m0 + ml) * 1024 + ncol] = pval;
            Pbuf[ml][ncol] = (_Float16)pval;
        }
    }
    __syncthreads();

    // ---- Phase 2: ctx = P @ V over this wave's K-slice [128w, 128w+128) ----
    v8f cacc[4];
#pragma unroll
    for (int t = 0; t < 4; ++t) cacc[t] = (v8f){};

#pragma unroll
    for (int j = 0; j < 4; ++j) {
        const int k0 = w * 128 + j * 32;
        V16H pa;   // A fragment from Pbuf, reused for all 4 d-tiles
        pa.h[0] = *reinterpret_cast<const v8h*>(&Pbuf[nlane][k0 + ka]);
        pa.h[1] = *reinterpret_cast<const v8h*>(&Pbuf[nlane][k0 + ka + 16]);
#pragma unroll
        for (int t = 0; t < 4; ++t) {
            const int d = t * 16 + nlane;
            const _Float16* vrow = vp + (size_t)d * 1024 + k0 + kb;
            V16H vb;
            vb.h[0] = *reinterpret_cast<const v8h*>(vrow);
            vb.h[1] = *reinterpret_cast<const v8h*>(vrow + 16);
            cacc[t] = __builtin_amdgcn_wmma_f32_16x16x32_f16(
                false, pa.v, false, vb.v, (short)0, cacc[t], false, false);
        }
    }

    // per-wave partials -> LDS, 8-way reduce, store merged-head ctx (f16)
#pragma unroll
    for (int t = 0; t < 4; ++t) {
#pragma unroll
        for (int r = 0; r < 8; ++r)
            ctxred[w][mbase + r][t * 16 + nlane] = cacc[t][r];
    }
    __syncthreads();
#pragma unroll
    for (int i = 0; i < 4; ++i) {
        const int idx  = tid + i * 256;      // 0..1023
        const int mrow = idx >> 6;
        const int d    = idx & 63;
        float s = 0.0f;
#pragma unroll
        for (int ww = 0; ww < 8; ++ww) s += ctxred[ww][mrow][d];
        ctx[((size_t)(b * 1024 + m0 + mrow)) * 1024 + h * 64 + d] = (_Float16)s;
    }
}

// ---------------- Output projection + bias + residual (f32 out) ------------
__global__ void __launch_bounds__(256)
mha_outproj_kernel(const _Float16* __restrict__ ctx, const _Float16* __restrict__ WO,
                   const float* __restrict__ bO, const float* __restrict__ resid,
                   float* __restrict__ out)
{
    const int m0 = blockIdx.y * 64;
    const int n0 = blockIdx.x * 128;
    v8f acc[2][2] = {};
    gemm_tile_f16x4(ctx, 1024, WO, 1024, 1024, m0, n0, acc);

    const int lane  = threadIdx.x & 31;
    const int wid   = threadIdx.x >> 5;
    const int mbase = (lane >> 4) << 3;
    const int nl    = lane & 15;
#pragma unroll
    for (int mi = 0; mi < 2; ++mi) {
#pragma unroll
        for (int ni = 0; ni < 2; ++ni) {
            const int n = n0 + (wid & 3) * 32 + ni * 16 + nl;
            const float bv = bO[n];
#pragma unroll
            for (int r = 0; r < 8; ++r) {
                const int m = m0 + (wid >> 2) * 32 + mi * 16 + mbase + r;
                out[(size_t)m * 1024 + n] =
                    acc[mi][ni][r] + bv + resid[(size_t)m * 1024 + n];
            }
        }
    }
}

// ---------------- LayerNorm, in place (1024 elems/row, eps=1e-5) -----------
__global__ void __launch_bounds__(256)
mha_layernorm_kernel(float* __restrict__ x, const float* __restrict__ gamma,
                     const float* __restrict__ beta)
{
    __shared__ float rsum[256];
    __shared__ float rsq[256];
    const int t = threadIdx.x;
    float* p = x + (size_t)blockIdx.x * 1024;
    float v0 = p[t], v1 = p[t + 256], v2 = p[t + 512], v3 = p[t + 768];
    rsum[t] = v0 + v1 + v2 + v3;
    rsq[t]  = v0 * v0 + v1 * v1 + v2 * v2 + v3 * v3;
    __syncthreads();
    for (int s = 128; s > 0; s >>= 1) {
        if (t < s) { rsum[t] += rsum[t + s]; rsq[t] += rsq[t + s]; }
        __syncthreads();
    }
    const float mean = rsum[0] * (1.0f / 1024.0f);
    const float var  = rsq[0] * (1.0f / 1024.0f) - mean * mean;
    const float rs   = rsqrtf(var + 1e-5f);
    p[t]       = (v0 - mean) * rs * gamma[t]       + beta[t];
    p[t + 256] = (v1 - mean) * rs * gamma[t + 256] + beta[t + 256];
    p[t + 512] = (v2 - mean) * rs * gamma[t + 512] + beta[t + 512];
    p[t + 768] = (v3 - mean) * rs * gamma[t + 768] + beta[t + 768];
}

extern "C" void kernel_launch(void* const* d_in, const int* in_sizes, int n_in,
                              void* d_out, int out_size, void* d_ws, size_t ws_size,
                              hipStream_t stream)
{
    (void)in_sizes; (void)n_in; (void)out_size; (void)ws_size;

    const float*         Query = (const float*)d_in[0];
    const float*         Key   = (const float*)d_in[1];
    const float*         Value = (const float*)d_in[2];
    const unsigned char* mask  = (const unsigned char*)d_in[3];
    const float* W_Q = (const float*)d_in[4];
    const float* b_Q = (const float*)d_in[5];
    const float* W_K = (const float*)d_in[6];
    const float* b_K = (const float*)d_in[7];
    const float* W_V = (const float*)d_in[8];
    const float* b_V = (const float*)d_in[9];
    const float* W_O = (const float*)d_in[10];
    const float* b_O = (const float*)d_in[11];
    const float* ln_gamma = (const float*)d_in[12];
    const float* ln_beta  = (const float*)d_in[13];

    float* out    = (float*)d_out;
    float* normed = out;                                   // [4,1024,1024]
    float* attn   = out + (size_t)4 * 1024 * 1024;         // [4,16,1024,1024]

    // workspace: 32M halfs = 64 MB, all f16
    _Float16* hs  = (_Float16*)d_ws;
    const size_t M4 = 4194304;       // 4M elements
    _Float16* qh  = hs;              // [b,h,s,d]
    _Float16* kh  = hs + 1 * M4;     // [b,h,s,d]
    _Float16* vh  = hs + 2 * M4;     // [b,h,d,s]
    _Float16* ctx = hs + 3 * M4;     // [b*s, h*d]
    _Float16* Qc  = hs + 4 * M4;     // converted activations
    _Float16* Kc  = hs + 5 * M4;
    _Float16* Vc  = hs + 6 * M4;
    _Float16* Wq  = hs + 7 * M4;     // converted weights (1M each)
    _Float16* Wk  = Wq + 1048576;
    _Float16* Wv  = Wk + 1048576;
    _Float16* Wo  = Wv + 1048576;

    const dim3 blk(256);

    // one-time f32 -> f16 conversions
    mha_cvt_f16_kernel<<<dim3(4096), blk, 0, stream>>>(Query, Qc, 4194304);
    mha_cvt_f16_kernel<<<dim3(4096), blk, 0, stream>>>(Key,   Kc, 4194304);
    mha_cvt_f16_kernel<<<dim3(4096), blk, 0, stream>>>(Value, Vc, 4194304);
    mha_cvt_f16_kernel<<<dim3(1024), blk, 0, stream>>>(W_Q, Wq, 1048576);
    mha_cvt_f16_kernel<<<dim3(1024), blk, 0, stream>>>(W_K, Wk, 1048576);
    mha_cvt_f16_kernel<<<dim3(1024), blk, 0, stream>>>(W_V, Wv, 1048576);
    mha_cvt_f16_kernel<<<dim3(1024), blk, 0, stream>>>(W_O, Wo, 1048576);

    // QKV projections (4096x1024x1024, f16 in/out, f32 accumulate)
    mha_proj_kernel<<<dim3(8, 64), blk, 0, stream>>>(Qc, Wq, b_Q, qh, 0);
    mha_proj_kernel<<<dim3(8, 64), blk, 0, stream>>>(Kc, Wk, b_K, kh, 0);
    mha_proj_kernel<<<dim3(8, 64), blk, 0, stream>>>(Vc, Wv, b_V, vh, 1);

    // fused scores + softmax + P@V
    mha_fused_attn_kernel<<<dim3(64, 64), blk, 0, stream>>>(qh, kh, vh, mask,
                                                            attn, ctx);

    // output projection + bias + residual (residual from original f32 Query)
    mha_outproj_kernel<<<dim3(8, 64), blk, 0, stream>>>(ctx, Wo, b_O, Query, normed);

    // LayerNorm in place
    mha_layernorm_kernel<<<dim3(4096), blk, 0, stream>>>(normed, ln_gamma, ln_beta);
}
